// SparseFusionLayer_67843303408387
// MI455X (gfx1250) — matrix operand order repaired and checked by reading, not compiled, not run on hardware
//
#include <hip/hip_runtime.h>

// ---------------------------------------------------------------------------
// Problem constants (from reference setup_inputs)
#define B_    4
#define L_    4096
#define N_    1024
#define DS    512     // D_SAT
#define DSOC  256     // D_SOC
#define H_    8
#define HD    64      // head dim
#define LN_EPS 1e-5f

typedef __attribute__((ext_vector_type(16))) __bf16 v16bf;
typedef __attribute__((ext_vector_type(8)))  __bf16 v8bf;
typedef __attribute__((ext_vector_type(8)))  float  v8f;
typedef __attribute__((ext_vector_type(8)))  unsigned short v8u16;

// ---------------------------------------------------------------------------
// bf16 helper (round-to-nearest-even) — one-shot conversions / epilogues only.
__device__ __forceinline__ unsigned short f2bfu(float f) {
  unsigned u = __builtin_bit_cast(unsigned, f);
  u = (u + 0x7FFFu + ((u >> 16) & 1u)) >> 16;
  return (unsigned short)u;
}

// ---------------------------------------------------------------------------
// WMMA wrapper: D = A(16x32 bf16) * B(32x16 bf16) + C(16x16 f32)
__device__ __forceinline__ v8f wmma_bf16(v16bf a, v16bf b, v8f c) {
  return __builtin_amdgcn_wmma_f32_16x16x32_bf16(false, a, false, b,
                                                 (short)0, c, false, false);
}

// A-fragment 16x32 bf16, row-major source.
// lanes 0-15: row=lane, K={0..7,16..23}; lanes 16-31: row=lane-16, K={8..15,24..31}
__device__ __forceinline__ v16bf load_a(const unsigned short* __restrict__ A,
                                        int lda, int lane) {
  const int row = lane & 15;
  const int klo = (lane & 16) ? 8 : 0;
  const unsigned short* p = A + (size_t)row * lda + klo;
  v8bf lo = __builtin_bit_cast(v8bf, *(const v8u16*)(p));
  v8bf hi = __builtin_bit_cast(v8bf, *(const v8u16*)(p + 16));
  v16bf a;
#pragma unroll
  for (int e = 0; e < 8; ++e) { a[e] = lo[e]; a[e + 8] = hi[e]; }
  return a;
}

// B-fragment 32x16 with B[k][n] = W[n][k]  (C = A @ W^T), W row-major bf16.
// lane = column n; lanes 0-15 hold K={0..7,16..23}, lanes 16-31 K={8..15,24..31}
// (same K order as load_a).
__device__ __forceinline__ v16bf load_bT(const unsigned short* __restrict__ W,
                                         int ldw, int lane) {
  const int col = lane & 15;
  const int kb  = (lane & 16) ? 8 : 0;
  const unsigned short* p = W + (size_t)col * ldw + kb;
  v8bf lo = __builtin_bit_cast(v8bf, *(const v8u16*)(p));
  v8bf hi = __builtin_bit_cast(v8bf, *(const v8u16*)(p + 16));
  v16bf b;
#pragma unroll
  for (int e = 0; e < 8; ++e) { b[e] = lo[e]; b[e + 8] = hi[e]; }
  return b;
}

// ---------------------------------------------------------------------------
// Kernel 0: one-shot f32 -> bf16 conversion (vectorized, memory bound)
__global__ __launch_bounds__(256)
void cvt_bf16_kernel(const float* __restrict__ X, unsigned short* __restrict__ Y,
                     int n4) {
  const int i = blockIdx.x * 256 + threadIdx.x;
  if (i < n4) {
    const float4 v = ((const float4*)X)[i];
    ushort4 o;
    o.x = f2bfu(v.x); o.y = f2bfu(v.y); o.z = f2bfu(v.z); o.w = f2bfu(v.w);
    ((ushort4*)Y)[i] = o;
  }
}

// ---------------------------------------------------------------------------
// Kernel 1: Y(bf16)[M,Dout] = A(bf16)[M,Din] @ W(bf16)[Dout,Din]^T + bias
// Block 256 = 8 waves; wave w -> rows (blockIdx.x*128 + w*16), 64 cols/block.
__global__ __launch_bounds__(256)
void gemm_bf16_kernel(const unsigned short* __restrict__ A,
                      const unsigned short* __restrict__ W,
                      const float* __restrict__ bias,
                      unsigned short* __restrict__ Y,
                      int Din, int Dout) {
  const int lane = threadIdx.x & 31;
  const int wave = threadIdx.x >> 5;
  const int r0 = blockIdx.x * 128 + wave * 16;
  const int c0 = blockIdx.y * 64;
  v8f vzero = {};
  v8f acc[4];
#pragma unroll
  for (int t = 0; t < 4; ++t) acc[t] = vzero;

  for (int k0 = 0; k0 < Din; k0 += 32) {
    v16bf a = load_a(A + (size_t)r0 * Din + k0, Din, lane);
#pragma unroll
    for (int t = 0; t < 4; ++t) {
      v16bf b = load_bT(W + (size_t)(c0 + t * 16) * Din + k0, Din, lane);
      acc[t] = wmma_bf16(a, b, acc[t]);
    }
  }
  const int rowb = (lane >> 4) << 3;
  const int col  = lane & 15;
#pragma unroll
  for (int t = 0; t < 4; ++t) {
    float bv = bias[c0 + t * 16 + col];
#pragma unroll
    for (int r = 0; r < 8; ++r)
      Y[(size_t)(r0 + rowb + r) * Dout + (c0 + t * 16 + col)] = f2bfu(acc[t][r] + bv);
  }
}

// ---------------------------------------------------------------------------
// Kernel 1b: V projection with transposed output:
//   Yt[b][c][key] = (social @ Wv^T + bv)[b*N_+key][c]   (Yt: [B][DS][N_] bf16)
// Rows r = b*N_ + key; 128-row tiles never cross a batch boundary (N_%128==0).
// Per-lane stores land contiguously along the key axis.
__global__ __launch_bounds__(256)
void gemm_bf16_vt_kernel(const unsigned short* __restrict__ A,
                         const unsigned short* __restrict__ W,
                         const float* __restrict__ bias,
                         unsigned short* __restrict__ Yt,
                         int Din) {
  const int lane = threadIdx.x & 31;
  const int wave = threadIdx.x >> 5;
  const int r0 = blockIdx.x * 128 + wave * 16;
  const int c0 = blockIdx.y * 64;
  v8f vzero = {};
  v8f acc[4];
#pragma unroll
  for (int t = 0; t < 4; ++t) acc[t] = vzero;

  for (int k0 = 0; k0 < Din; k0 += 32) {
    v16bf a = load_a(A + (size_t)r0 * Din + k0, Din, lane);
#pragma unroll
    for (int t = 0; t < 4; ++t) {
      v16bf b = load_bT(W + (size_t)(c0 + t * 16) * Din + k0, Din, lane);
      acc[t] = wmma_bf16(a, b, acc[t]);
    }
  }
  const int rowb = (lane >> 4) << 3;
  const int col  = lane & 15;
  const int bb   = r0 / N_;
  const int key0 = (r0 % N_) + rowb;
#pragma unroll
  for (int t = 0; t < 4; ++t) {
    const int c = c0 + t * 16 + col;
    const float bv = bias[c];
    unsigned short* p = Yt + ((size_t)bb * DS + c) * N_ + key0;
#pragma unroll
    for (int r = 0; r < 8; ++r) p[r] = f2bfu(acc[t][r] + bv);
  }
}

// ---------------------------------------------------------------------------
// Kernel 2: flash cross-attention. grid = (L/128, B*H); 8 waves x 16 rows.
// K and V^T tiles for each 32-key block are staged cooperatively in LDS once
// per block (one b128 per thread), then all 8 waves build WMMA fragments from
// LDS — removes the 8x redundant global fragment loads.
__global__ __launch_bounds__(256)
void attn_kernel(const unsigned short* __restrict__ Qb,
                 const unsigned short* __restrict__ Kb,
                 const unsigned short* __restrict__ Vt,  // [B][DS][N_]
                 const int* __restrict__ mask0,          // mask[0]: (L, N)
                 unsigned short* __restrict__ Ctx) {
  __shared__ unsigned short Ktile[32 * 64];     // [key][dim]
  __shared__ unsigned short Vtile[64 * 32];     // [dim][key]
  __shared__ unsigned short ldsP[8][16 * 32];   // per-wave P tiles
  const int tid  = threadIdx.x;
  const int lane = tid & 31;
  const int wave = tid >> 5;
  const int q0 = blockIdx.x * 128 + wave * 16;
  const int b  = blockIdx.y >> 3;
  const int h  = blockIdx.y & 7;
  const int rowb = (lane >> 4) << 3;
  const int col  = lane & 15;
  // staging coordinates: 16 bytes (8 bf16) per thread per tile
  const int kkey = tid >> 3, kdim = (tid & 7) * 8;   // K tile
  const int vdim = tid >> 2, vkey = (tid & 3) * 8;   // V^T tile

  const unsigned short* Qp = Qb + ((size_t)(b * L_ + q0) * DS + h * HD);
  const v16bf aq0 = load_a(Qp, DS, lane);        // k = 0..31
  const v16bf aq1 = load_a(Qp + 32, DS, lane);   // k = 32..63

  v8f vzero = {};
  v8f acc[4];
#pragma unroll
  for (int t = 0; t < 4; ++t) acc[t] = vzero;
  float mr[8], lr[8];
#pragma unroll
  for (int r = 0; r < 8; ++r) { mr[r] = -3.0e38f; lr[r] = 0.0f; }

  const unsigned short* Kg = Kb + ((size_t)(b * N_ + kkey) * DS + h * HD + kdim);
  const unsigned short* Vg = Vt + ((size_t)(b * DS + h * HD + vdim)) * N_ + vkey;

  for (int n0 = 0; n0 < N_; n0 += 32) {
    if (n0) __syncthreads();           // previous block's tiles fully consumed
    *(v8u16*)&Ktile[kkey * 64 + kdim] = *(const v8u16*)(Kg + (size_t)n0 * DS);
    *(v8u16*)&Vtile[vdim * 32 + vkey] = *(const v8u16*)(Vg + n0);
    if (n0 + 32 < N_) {                // global_prefetch_b8 next block
      __builtin_prefetch(Kg + (size_t)(n0 + 32) * DS, 0, 1);
      __builtin_prefetch(Vg + n0 + 32, 0, 1);
    }
    __syncthreads();

    // S = Q @ K^T : two 16x16 key tiles, K-dim = 64 (fragments from LDS)
    v8f s0 = vzero, s1 = vzero;
    s0 = wmma_bf16(aq0, load_bT(Ktile, 64, lane), s0);
    s0 = wmma_bf16(aq1, load_bT(Ktile + 32, 64, lane), s0);
    s1 = wmma_bf16(aq0, load_bT(Ktile + 16 * 64, 64, lane), s1);
    s1 = wmma_bf16(aq1, load_bT(Ktile + 16 * 64 + 32, 64, lane), s1);

    // scale + mask bias (mask[0] shared across batch, per reference quirk)
#pragma unroll
    for (int r = 0; r < 8; ++r) {
      const int l = q0 + rowb + r;
      float b0 = mask0[(size_t)l * N_ + n0 + col]      ? 0.0f : -1e9f;
      float b1 = mask0[(size_t)l * N_ + n0 + 16 + col] ? 0.0f : -1e9f;
      s0[r] = s0[r] * 0.125f + b0;   // 1/sqrt(64)
      s1[r] = s1[r] * 0.125f + b1;
    }
    // online softmax: each row lives across a 16-lane group
#pragma unroll
    for (int r = 0; r < 8; ++r) {
      float mx = fmaxf(s0[r], s1[r]);
#pragma unroll
      for (int off = 1; off < 16; off <<= 1) mx = fmaxf(mx, __shfl_xor(mx, off, 32));
      mx = fmaxf(mx, mr[r]);
      const float corr = __expf(mr[r] - mx);
      const float p0 = __expf(s0[r] - mx);
      const float p1 = __expf(s1[r] - mx);
      s0[r] = p0; s1[r] = p1;
      float rs = p0 + p1;
#pragma unroll
      for (int off = 1; off < 16; off <<= 1) rs += __shfl_xor(rs, off, 32);
      lr[r] = lr[r] * corr + rs;
      mr[r] = mx;
#pragma unroll
      for (int t = 0; t < 4; ++t) acc[t][r] *= corr;
    }
    // P: D-layout -> A-fragment layout via per-wave LDS tile
    unsigned short* Pt = &ldsP[wave][0];
#pragma unroll
    for (int r = 0; r < 8; ++r) {
      Pt[(rowb + r) * 32 + col]      = f2bfu(s0[r]);
      Pt[(rowb + r) * 32 + 16 + col] = f2bfu(s1[r]);
    }
    asm volatile("s_wait_dscnt 0" ::: "memory");  // wave-private tile: DS RAW fence
    v16bf ap = load_a(Pt, 32, lane);
    // O += P(16x32) @ V(32x64): B[k][n] = Vtile[n][k] -> contiguous LDS runs
#pragma unroll
    for (int t = 0; t < 4; ++t)
      acc[t] = wmma_bf16(ap, load_bT(Vtile + t * 16 * 32, 32, lane), acc[t]);
  }

  unsigned short* Cp = Ctx + ((size_t)(b * L_ + q0) * DS + h * HD);
#pragma unroll
  for (int r = 0; r < 8; ++r) {
    const float inv = 1.0f / lr[r];
#pragma unroll
    for (int t = 0; t < 4; ++t)
      Cp[(size_t)(rowb + r) * DS + t * 16 + col] = f2bfu(acc[t][r] * inv);
  }
}

// ---------------------------------------------------------------------------
// Kernel 3: attn_out = Ctx(bf16) @ Wo(bf16)^T + bo + residual(sat f32), f32 out
__global__ __launch_bounds__(256)
void outproj_res_kernel(const unsigned short* __restrict__ Ctx,
                        const unsigned short* __restrict__ Wo,
                        const float* __restrict__ bo,
                        const float* __restrict__ X0, float* __restrict__ Y) {
  const int lane = threadIdx.x & 31;
  const int wave = threadIdx.x >> 5;
  const int r0 = blockIdx.x * 128 + wave * 16;
  const int c0 = blockIdx.y * 64;
  v8f vzero = {};
  v8f acc[4];
#pragma unroll
  for (int t = 0; t < 4; ++t) acc[t] = vzero;

  for (int k0 = 0; k0 < DS; k0 += 32) {
    v16bf a = load_a(Ctx + (size_t)r0 * DS + k0, DS, lane);
#pragma unroll
    for (int t = 0; t < 4; ++t) {
      v16bf b = load_bT(Wo + (size_t)(c0 + t * 16) * DS + k0, DS, lane);
      acc[t] = wmma_bf16(a, b, acc[t]);
    }
  }
  const int rowb = (lane >> 4) << 3;
  const int col  = lane & 15;
#pragma unroll
  for (int t = 0; t < 4; ++t) {
    const int c = c0 + t * 16 + col;
    const float bv = bo[c];
#pragma unroll
    for (int r = 0; r < 8; ++r) {
      const size_t idx = (size_t)(r0 + rowb + r) * DS + c;
      Y[idx] = acc[t][r] + bv + X0[idx];
    }
  }
}

// ---------------------------------------------------------------------------
// Kernel 4: row LayerNorm over 512 elements; one block per row.
__global__ __launch_bounds__(256)
void ln_kernel(const float* __restrict__ X, const float* __restrict__ gamma,
               const float* __restrict__ beta, float* __restrict__ Out) {
  __shared__ float sbuf[8];
  const size_t row = blockIdx.x;
  const int t = threadIdx.x;
  const int lane = t & 31, wave = t >> 5;
  const float* x = X + row * DS;
  const float v0 = x[t], v1 = x[t + 256];

  float s = v0 + v1;
#pragma unroll
  for (int off = 1; off < 32; off <<= 1) s += __shfl_xor(s, off, 32);
  if (lane == 0) sbuf[wave] = s;
  __syncthreads();
  float mu = 0.0f;
#pragma unroll
  for (int i = 0; i < 8; ++i) mu += sbuf[i];
  mu *= (1.0f / (float)DS);
  __syncthreads();

  const float d0 = v0 - mu, d1 = v1 - mu;
  s = d0 * d0 + d1 * d1;
#pragma unroll
  for (int off = 1; off < 32; off <<= 1) s += __shfl_xor(s, off, 32);
  if (lane == 0) sbuf[wave] = s;
  __syncthreads();
  float var = 0.0f;
#pragma unroll
  for (int i = 0; i < 8; ++i) var += sbuf[i];
  var *= (1.0f / (float)DS);
  const float rinv = rsqrtf(var + LN_EPS);

  Out[row * DS + t]       = d0 * rinv * gamma[t] + beta[t];
  Out[row * DS + t + 256] = d1 * rinv * gamma[t + 256] + beta[t + 256];
}

// ---------------------------------------------------------------------------
extern "C" void kernel_launch(void* const* d_in, const int* in_sizes, int n_in,
                              void* d_out, int out_size, void* d_ws, size_t ws_size,
                              hipStream_t stream) {
  (void)in_sizes; (void)n_in; (void)out_size; (void)ws_size;
  const float* sat   = (const float*)d_in[0];
  const float* soc   = (const float*)d_in[1];
  const int*   mask  = (const int*)d_in[2];    // mask[0] slice = first L*N ints
  const float* Wq    = (const float*)d_in[3];
  const float* bq    = (const float*)d_in[4];
  const float* Wk    = (const float*)d_in[5];
  const float* bk    = (const float*)d_in[6];
  const float* Wv    = (const float*)d_in[7];
  const float* bv    = (const float*)d_in[8];
  const float* Wo    = (const float*)d_in[9];
  const float* bo    = (const float*)d_in[10];
  const float* gamma = (const float*)d_in[11];
  const float* beta  = (const float*)d_in[12];
  float* out = (float*)d_out;

  const size_t M1 = (size_t)B_ * L_;   // 16384 query rows
  const size_t M2 = (size_t)B_ * N_;   // 4096 key/value rows

  // ---- workspace carve-up (bytes), 16B-aligned sections ----
  char* ws = (char*)d_ws;
  unsigned short* Qb   = (unsigned short*)(ws);                       size_t o = M1 * DS * 2;
  unsigned short* Kb   = (unsigned short*)(ws + o);                   o += M2 * DS * 2;
  unsigned short* Vtb  = (unsigned short*)(ws + o);                   o += M2 * DS * 2;  // [B][DS][N]
  unsigned short* Ctx  = (unsigned short*)(ws + o);                   o += M1 * DS * 2;
  unsigned short* Wqb  = (unsigned short*)(ws + o);                   o += (size_t)DS * DS * 2;
  unsigned short* Wkb  = (unsigned short*)(ws + o);                   o += (size_t)DS * DSOC * 2;
  unsigned short* Wvb  = (unsigned short*)(ws + o);                   o += (size_t)DS * DSOC * 2;
  unsigned short* Wob  = (unsigned short*)(ws + o);                   o += (size_t)DS * DS * 2;
  // staging region reused: satb/socb live first, AOut aliases it afterwards
  unsigned short* satb = (unsigned short*)(ws + o);
  unsigned short* socb = satb + M1 * DS;
  float*          AOut = (float*)(ws + o);   // reuses satb/socb region (33.6 MB)

  // ---- one-shot f32 -> bf16 conversions (memory bound) ----
  const int nsat = (int)(M1 * DS / 4), nsoc = (int)(M2 * DSOC / 4);
  const int nwqo = DS * DS / 4, nwkv = DS * DSOC / 4;
  cvt_bf16_kernel<<<(nsat + 255) / 256, 256, 0, stream>>>(sat, satb, nsat);
  cvt_bf16_kernel<<<(nsoc + 255) / 256, 256, 0, stream>>>(soc, socb, nsoc);
  cvt_bf16_kernel<<<(nwqo + 255) / 256, 256, 0, stream>>>(Wq, Wqb, nwqo);
  cvt_bf16_kernel<<<(nwkv + 255) / 256, 256, 0, stream>>>(Wk, Wkb, nwkv);
  cvt_bf16_kernel<<<(nwkv + 255) / 256, 256, 0, stream>>>(Wv, Wvb, nwkv);
  cvt_bf16_kernel<<<(nwqo + 255) / 256, 256, 0, stream>>>(Wo, Wob, nwqo);

  // ---- Q/K projections (row-major out), V projection (transposed out) ----
  gemm_bf16_kernel<<<dim3((unsigned)(M1 / 128), DS / 64), 256, 0, stream>>>(satb, Wqb, bq, Qb, DS, DS);
  gemm_bf16_kernel<<<dim3((unsigned)(M2 / 128), DS / 64), 256, 0, stream>>>(socb, Wkb, bk, Kb, DSOC, DS);
  gemm_bf16_vt_kernel<<<dim3((unsigned)(M2 / 128), DS / 64), 256, 0, stream>>>(socb, Wvb, bv, Vtb, DSOC);
  // ---- Flash cross-attention (LDS-staged K / V^T tiles) ----
  attn_kernel<<<dim3(L_ / 128, B_ * H_), 256, 0, stream>>>(Qb, Kb, Vtb, mask, Ctx);
  // ---- Output projection + residual (overwrites staging region) ----
  outproj_res_kernel<<<dim3((unsigned)(M1 / 128), DS / 64), 256, 0, stream>>>(Ctx, Wob, bo, sat, AOut);
  // ---- LayerNorm ----
  ln_kernel<<<(unsigned)M1, 256, 0, stream>>>(AOut, gamma, beta, out);
}
